// Quantized_506806141434
// MI455X (gfx1250) — matrix operand order repaired and checked by reading, not compiled, not run on hardware
//
#include <hip/hip_runtime.h>

typedef __attribute__((ext_vector_type(16))) _Float16 v16h;
typedef __attribute__((ext_vector_type(8)))  _Float16 v8h;
typedef __attribute__((ext_vector_type(8)))  float    v8f;

#define BATCH 16
#define NDIM  8192
#define MDIM  32768

// ---------------------------------------------------------------------------
// Prep: y fp32 -> f16 (256KB, stays hot in the 192MB L2 for the main pass)
// plus per-row sums ysum[b].
// ---------------------------------------------------------------------------
__global__ void __launch_bounds__(256) prep_y_kernel(
    const float* __restrict__ y, _Float16* __restrict__ y16,
    float* __restrict__ ysum) {
  __shared__ float red[256];
  const int b = blockIdx.x;     // 0..15 (batch row)
  const int t = threadIdx.x;    // 0..255
  const float* yr = y   + (size_t)b * NDIM;
  _Float16*    dr = y16 + (size_t)b * NDIM;
  float s = 0.f;
  for (int i = t; i < NDIM; i += 256) {
    float v = yr[i];
    s += v;
    dr[i] = (_Float16)v;
  }
  red[t] = s;
  __syncthreads();
  for (int off = 128; off > 0; off >>= 1) {
    if (t < off) red[t] += red[t + off];
    __syncthreads();
  }
  if (t == 0) ysum[b] = red[0];
}

// ---------------------------------------------------------------------------
// Main: split-K pair of wave32s per 16x16 output tile. Each wave streams its
// half of the 1 GiB int32 weight exactly once (coalesced b32 rows), converts
// u8 payload -> f16 in-register (v_cvt_f16_u16), and feeds
// v_wmma_f32_16x16x32_f16. K-halves are combined through LDS, then the
// even wave applies mrange/offset/ysum and stores.
// 512 blocks x 8 waves = 4096 waves; unroll-4 keeps ~63 loads in flight/wave.
// ---------------------------------------------------------------------------
__global__ void __launch_bounds__(256) mm8_wmma_kernel(
    const int* __restrict__ w,          // [N, M] int32, values 0..255
    const _Float16* __restrict__ y16,   // [B, N] f16 (prepped)
    const float* __restrict__ ysum,     // [B]
    const float* __restrict__ mrange,   // [M]
    const float* __restrict__ offs,     // [M]
    float* __restrict__ out) {          // [B, M] fp32
  __shared__ float partial[4][32][8];   // [tile][lane][acc reg]

  const int lane  = threadIdx.x & 31;
  const int wave  = threadIdx.x >> 5;   // 0..7
  const int tile  = wave >> 1;          // 0..3: which 16-column tile
  const int khalf = wave & 1;           // 0/1: which K half
  const int m0    = (blockIdx.x * 4 + tile) << 4;
  const int col   = lane & 15;
  const int hi    = lane >> 4;          // lane half selects K sub-range
  const int bK    = hi << 4;            // B fragment: K base 0 or 16
  const int aK    = hi << 4;            // A fragment: K base 0 or 16
  const int kbeg  = khalf * (NDIM / 2);

  // B: lane j holds column m0+j, 16 consecutive K rows (2 f16 / VGPR).
  const int* wp = w + (size_t)(kbeg + bK) * MDIM + m0 + col;
  // A: lane i holds y row i, 16 consecutive K [aK..aK+15].
  const _Float16* ap = y16 + (size_t)col * NDIM + kbeg + aK;

  v8f acc = {};
#pragma unroll 4
  for (int k0 = 0; k0 < NDIM / 2; k0 += 32) {
    // ---- A fragment: two aligned 16B loads of pre-converted f16 y ----
    v8h alo = *(const v8h*)(ap);
    v8h ahi = *(const v8h*)(ap + 8);
    v16h a;
#pragma unroll
    for (int i = 0; i < 8; ++i) { a[i] = alo[i]; a[8 + i] = ahi[i]; }

    // ---- B fragment: 16 coalesced b32 loads; u16->f16 convert is exact ----
    v16h bm;
#pragma unroll
    for (int r = 0; r < 16; ++r)
      bm[r] = (_Float16)(unsigned short)wp[(size_t)r * MDIM];

    acc = __builtin_amdgcn_wmma_f32_16x16x32_f16(
        /*neg_a=*/false, a, /*neg_b=*/false, bm,
        /*c_mod=*/(short)0, acc, /*reuse_a=*/false, /*reuse_b=*/false);

    wp += (size_t)32 * MDIM;
    ap += 32;
  }

  // ---- Combine the two K halves through LDS (wave-uniform branches) ----
  if (khalf) {
#pragma unroll
    for (int v = 0; v < 8; ++v) partial[tile][lane][v] = acc[v];
  }
  __syncthreads();
  if (!khalf) {
#pragma unroll
    for (int v = 0; v < 8; ++v) acc[v] += partial[tile][lane][v];

    // ---- Epilogue: D layout is VGPR v -> row (v + 8*hi), N = lane&15 ----
    const float mr = mrange[m0 + col];
    const float of = offs[m0 + col];
    const int rowbase = hi << 3;
    float* op = out + m0 + col;
#pragma unroll
    for (int v = 0; v < 8; ++v) {
      const int row = rowbase + v;
      op[(size_t)row * MDIM] = mr * acc[v] + ysum[row] * of;
    }
  }
}

// ---------------------------------------------------------------------------
extern "C" void kernel_launch(void* const* d_in, const int* in_sizes, int n_in,
                              void* d_out, int out_size, void* d_ws,
                              size_t ws_size, hipStream_t stream) {
  const float* y      = (const float*)d_in[0];
  const int*   w      = (const int*)d_in[1];
  const float* mrange = (const float*)d_in[2];
  const float* offs   = (const float*)d_in[3];
  float*       out    = (float*)d_out;

  _Float16* y16  = (_Float16*)d_ws;
  float*    ysum = (float*)((char*)d_ws + (size_t)BATCH * NDIM * sizeof(_Float16));

  prep_y_kernel<<<BATCH, 256, 0, stream>>>(y, y16, ysum);

  // 32768 columns / (16 cols/tile * 4 tiles/block) = 512 blocks, 8 waves each.
  mm8_wmma_kernel<<<MDIM / (16 * 4), 256, 0, stream>>>(w, y16, ysum, mrange,
                                                       offs, out);
}
